// Attention_52218212385160
// MI455X (gfx1250) — matrix operand order repaired and checked by reading, not compiled, not run on hardware
//
#include <hip/hip_runtime.h>
#include <hip/hip_bf16.h>

// ---------------------------------------------------------------------------
// Causal MHA forward for MI455X (gfx1250, wave32, WMMA).
// All GEMMs use v_wmma_f32_16x16x32_f16 (f16 A/B, f32 accumulate).
// ---------------------------------------------------------------------------

#define BATCH   4
#define SEQ     2048
#define HEADS   16
#define DMODEL  1024
#define DHEAD   64
#define ROWS    (BATCH * SEQ)          // 8192

typedef __attribute__((ext_vector_type(16))) _Float16 v16h;
typedef __attribute__((ext_vector_type(8)))  float    v8f;

union F16Frag {
    v16h h;
    unsigned int u[8];
};

__device__ __forceinline__ v8f wmma_f16(v16h a, v16h b, v8f c) {
    return __builtin_amdgcn_wmma_f32_16x16x32_f16(
        /*neg_a=*/false, a, /*neg_b=*/false, b,
        /*c_mod=*/(short)0, c, /*reuse_a=*/false, /*reuse_b=*/false);
}

__device__ __forceinline__ v8f zero8() {
    v8f z;
#pragma unroll
    for (int i = 0; i < 8; ++i) z[i] = 0.0f;
    return z;
}

// A-matrix 16x32 f16 fragment. base_row = start of this lane's row (m = lane&15)
// at the tile's K offset 0. ISA K-pair order: VGPR i<4 -> K=8*half+2i,
// VGPR i>=4 -> K=16+8*half+2(i-4). All pair offsets even -> 4B-aligned u32 loads.
__device__ __forceinline__ v16h load_a_frag(const _Float16* base_row, int half) {
    F16Frag f;
#pragma unroll
    for (int i = 0; i < 8; ++i) {
        int kp = (i < 4) ? (8 * half + 2 * i) : (16 + 8 * half + 2 * (i - 4));
        f.u[i] = *(const unsigned int*)(base_row + kp);
    }
    return f.h;
}

// B-matrix 32x16 f16 fragment. base_col = this lane's column data (n = lane&15),
// K-contiguous. Same K-pair order as A: VGPR i<4 -> K=8*half+2i,
// VGPR i>=4 -> K=16+8*half+2(i-4).
__device__ __forceinline__ v16h load_b_frag(const _Float16* base_col, int half) {
    F16Frag f;
#pragma unroll
    for (int i = 0; i < 8; ++i) {
        int kp = (i < 4) ? (8 * half + 2 * i) : (16 + 8 * half + 2 * (i - 4));
        f.u[i] = *(const unsigned int*)(base_col + kp);
    }
    return f.h;
}

// ---------------------------------------------------------------------------
// Kernel 0: f32 -> f16 casts + weight transposes.
//   xh  [8192][1024]          = x
//   wqT/wkT/wvT [h][64][1024] = W_*[h][m][d] transposed (d-major)
//   woT [1024][1024]          = W_O[(h,d)][m] transposed (m-major)
// ---------------------------------------------------------------------------
__global__ void cast_kernel(const float* __restrict__ x,
                            const float* __restrict__ WQ,
                            const float* __restrict__ WK,
                            const float* __restrict__ WV,
                            const float* __restrict__ WO,
                            _Float16* __restrict__ xh,
                            _Float16* __restrict__ wqT,
                            _Float16* __restrict__ wkT,
                            _Float16* __restrict__ wvT,
                            _Float16* __restrict__ woT) {
    const int tid = blockIdx.x * blockDim.x + threadIdx.x;
    const int nt  = gridDim.x * blockDim.x;
    for (int i = tid; i < ROWS * DMODEL; i += nt)
        xh[i] = (_Float16)x[i];
    for (int i = tid; i < HEADS * DMODEL * DHEAD; i += nt) {
        int h   = i >> 16;          // / (1024*64)
        int rem = i & 65535;
        int m   = rem >> 6;         // d_model index
        int d   = rem & 63;         // d_head index
        int dst = (h << 16) + (d << 10) + m;
        wqT[dst] = (_Float16)WQ[i];
        wkT[dst] = (_Float16)WK[i];
        wvT[dst] = (_Float16)WV[i];
    }
    for (int i = tid; i < DMODEL * DMODEL; i += nt) {
        int hd = i >> 10;           // (h,d) row of flat W_O
        int mm = i & 1023;          // d_model col
        woT[mm * DMODEL + hd] = (_Float16)WO[i];
    }
}

// ---------------------------------------------------------------------------
// Kernel 1: QKV projection. One wave computes a 16x64 tile of q, k, or v for
// one head. q,k stored [b][h][s][64]; v stored transposed [b][h][64][s].
// ---------------------------------------------------------------------------
__global__ void qkv_kernel(const _Float16* __restrict__ xh,
                           const _Float16* __restrict__ wqT,
                           const _Float16* __restrict__ wkT,
                           const _Float16* __restrict__ wvT,
                           const float* __restrict__ bQ,
                           const float* __restrict__ bK,
                           const float* __restrict__ bV,
                           _Float16* __restrict__ q,
                           _Float16* __restrict__ k,
                           _Float16* __restrict__ vT) {
    const int lane = threadIdx.x & 31;
    const int wv   = (blockIdx.x * blockDim.x + threadIdx.x) >> 5;
    const int mat  = wv % 3;
    const int head = (wv / 3) & (HEADS - 1);
    const int rt   = wv / (3 * HEADS);           // 0..511
    const int n    = lane & 15;
    const int half = lane >> 4;

    const _Float16* wT = (mat == 0) ? wqT : (mat == 1) ? wkT : wvT;
    wT += (size_t)head * DHEAD * DMODEL;
    const float* bias = ((mat == 0) ? bQ : (mat == 1) ? bK : bV) + head * DHEAD;

    const int r0 = rt * 16;                      // global row (b*s)
    const _Float16* arow = xh + (size_t)(r0 + n) * DMODEL;

    v8f acc[4];
#pragma unroll
    for (int ct = 0; ct < 4; ++ct) acc[ct] = zero8();

    for (int kk = 0; kk < DMODEL; kk += 32) {
        v16h a = load_a_frag(arow + kk, half);
#pragma unroll
        for (int ct = 0; ct < 4; ++ct) {
            v16h b = load_b_frag(wT + (size_t)(ct * 16 + n) * DMODEL + kk, half);
            acc[ct] = wmma_f16(a, b, acc[ct]);
        }
    }

    const int b_ = r0 / SEQ;
    const int s0 = r0 % SEQ;
    if (mat < 2) {
        _Float16* outp = ((mat == 0) ? q : k) +
                         (((size_t)b_ * HEADS + head) * SEQ + s0) * DHEAD;
#pragma unroll
        for (int ct = 0; ct < 4; ++ct) {
            float bb = bias[ct * 16 + n];
#pragma unroll
            for (int r = 0; r < 8; ++r) {
                int m2 = 8 * half + r;
                outp[(size_t)m2 * DHEAD + ct * 16 + n] = (_Float16)(acc[ct][r] + bb);
            }
        }
    } else {
        _Float16* outp = vT + (((size_t)b_ * HEADS + head) * DHEAD) * SEQ;
#pragma unroll
        for (int ct = 0; ct < 4; ++ct) {
            float bb = bias[ct * 16 + n];
#pragma unroll
            for (int r = 0; r < 8; ++r) {
                int m2 = 8 * half + r;
                outp[(size_t)(ct * 16 + n) * SEQ + s0 + m2] = (_Float16)(acc[ct][r] + bb);
            }
        }
    }
}

// ---------------------------------------------------------------------------
// Kernel 2: flash attention. One wave per (b, h, 16-query tile). Iterates
// 32-key blocks up to the causal diagonal with online softmax. P tile is
// re-laid-out C-fragment -> A-fragment through LDS. z stored [b][s][h*64] f16.
// ---------------------------------------------------------------------------
__global__ void attn_kernel(const _Float16* __restrict__ q,
                            const _Float16* __restrict__ k,
                            const _Float16* __restrict__ vT,
                            _Float16* __restrict__ z) {
    __shared__ _Float16 Plds[4][16 * 32];

    const int lane = threadIdx.x & 31;
    const int wid  = threadIdx.x >> 5;
    const int gw   = blockIdx.x * 4 + wid;
    const int b_   = gw / (HEADS * (SEQ / 16));
    const int rem  = gw % (HEADS * (SEQ / 16));
    const int h    = rem / (SEQ / 16);
    const int qt   = rem % (SEQ / 16);
    const int q0   = qt * 16;
    const int n    = lane & 15;
    const int half = lane >> 4;

    const _Float16* qb  = q  + (((size_t)b_ * HEADS + h) * SEQ) * DHEAD;
    const _Float16* kb_ = k  + (((size_t)b_ * HEADS + h) * SEQ) * DHEAD;
    const _Float16* vtb = vT + (((size_t)b_ * HEADS + h) * DHEAD) * SEQ;
    _Float16* pl = &Plds[wid][0];

    // Q tile A-fragments (rows m = lane&15), d halves 0..31 and 32..63.
    const _Float16* qrow = qb + (size_t)(q0 + n) * DHEAD;
    const v16h aq0 = load_a_frag(qrow,      half);
    const v16h aq1 = load_a_frag(qrow + 32, half);

    float mrow[8], lsum[8];
#pragma unroll
    for (int r = 0; r < 8; ++r) { mrow[r] = -1.0e30f; lsum[r] = 0.0f; }
    v8f o[4];
#pragma unroll
    for (int ct = 0; ct < 4; ++ct) o[ct] = zero8();

    for (int kb0 = 0; kb0 < q0 + 16; kb0 += 32) {
        // ---- scores: two 16x16 subtiles over keys kb0..kb0+31 ----
        v8f sc[2];
#pragma unroll
        for (int t2 = 0; t2 < 2; ++t2) {
            const int kt = kb0 + 16 * t2;
            const _Float16* krow = kb_ + (size_t)(kt + n) * DHEAD;
            v8f s = wmma_f16(aq0, load_b_frag(krow,      half), zero8());
            s     = wmma_f16(aq1, load_b_frag(krow + 32, half), s);
            const int key = kt + n;
#pragma unroll
            for (int r = 0; r < 8; ++r) {
                float v = s[r] * 0.125f;                 // 1/sqrt(64)
                if (key > q0 + 8 * half + r) v = -1.0e30f;  // causal mask
                s[r] = v;
            }
            sc[t2] = s;
        }
        // ---- row max across 16 lanes (butterfly) ----
        float bm[8];
#pragma unroll
        for (int r = 0; r < 8; ++r) {
            float v = fmaxf(sc[0][r], sc[1][r]);
            v = fmaxf(v, __shfl_xor(v, 1, 32));
            v = fmaxf(v, __shfl_xor(v, 2, 32));
            v = fmaxf(v, __shfl_xor(v, 4, 32));
            v = fmaxf(v, __shfl_xor(v, 8, 32));
            bm[r] = v;
        }
        float alpha[8], bsum[8];
#pragma unroll
        for (int r = 0; r < 8; ++r) {
            float mn = fmaxf(mrow[r], bm[r]);
            alpha[r] = __expf(mrow[r] - mn);
            mrow[r]  = mn;
            float p0 = __expf(sc[0][r] - mn);
            float p1 = __expf(sc[1][r] - mn);
            sc[0][r] = p0;
            sc[1][r] = p1;
            bsum[r]  = p0 + p1;
        }
#pragma unroll
        for (int r = 0; r < 8; ++r) {
            float v = bsum[r];
            v += __shfl_xor(v, 1, 32);
            v += __shfl_xor(v, 2, 32);
            v += __shfl_xor(v, 4, 32);
            v += __shfl_xor(v, 8, 32);
            lsum[r] = lsum[r] * alpha[r] + v;
        }
#pragma unroll
        for (int ct = 0; ct < 4; ++ct)
#pragma unroll
            for (int r = 0; r < 8; ++r) o[ct][r] *= alpha[r];

        // ---- P (C layout) -> LDS [16 q][32 keys] f16 ----
#pragma unroll
        for (int t2 = 0; t2 < 2; ++t2)
#pragma unroll
            for (int r = 0; r < 8; ++r)
                pl[(8 * half + r) * 32 + t2 * 16 + n] = (_Float16)(sc[t2][r]);
        // wave-lockstep: DS ops from one wave stay in order.

        // ---- P A-fragment + V B-fragments; O += P x V ----
        v16h ap = load_a_frag(pl + (size_t)n * 32, half);
#pragma unroll
        for (int ct = 0; ct < 4; ++ct) {
            const _Float16* vcol = vtb + (size_t)(ct * 16 + n) * SEQ + kb0;
            o[ct] = wmma_f16(ap, load_b_frag(vcol, half), o[ct]);
        }
    }

    // ---- epilogue: normalize and store z [b][s][h*64] ----
    const size_t zbase = ((size_t)b_ * SEQ + q0) * DMODEL + h * DHEAD;
#pragma unroll
    for (int r = 0; r < 8; ++r) {
        float inv = 1.0f / lsum[r];
#pragma unroll
        for (int ct = 0; ct < 4; ++ct)
            z[zbase + (size_t)(8 * half + r) * DMODEL + ct * 16 + n] =
                (_Float16)(o[ct][r] * inv);
    }
}

// ---------------------------------------------------------------------------
// Kernel 3: output projection. out[8192][1024] = z x WoT + b_O (f32 out).
// One wave per 16x16 output tile.
// ---------------------------------------------------------------------------
__global__ void oproj_kernel(const _Float16* __restrict__ z,
                             const _Float16* __restrict__ woT,
                             const float* __restrict__ bO,
                             float* __restrict__ out) {
    const int lane  = threadIdx.x & 31;
    const int gw    = (blockIdx.x * blockDim.x + threadIdx.x) >> 5;
    const int rt    = gw / (DMODEL / 16);
    const int ctile = gw % (DMODEL / 16);
    const int n     = lane & 15;
    const int half  = lane >> 4;

    const _Float16* arow = z   + (size_t)(rt * 16 + n) * DMODEL;
    const _Float16* bcol = woT + (size_t)(ctile * 16 + n) * DMODEL;

    v8f acc = zero8();
    for (int kk = 0; kk < DMODEL; kk += 32) {
        v16h a = load_a_frag(arow + kk, half);
        v16h b = load_b_frag(bcol + kk, half);
        acc = wmma_f16(a, b, acc);
    }
    const int col = ctile * 16 + n;
    const float bb = bO[col];
#pragma unroll
    for (int r = 0; r < 8; ++r)
        out[(size_t)(rt * 16 + 8 * half + r) * DMODEL + col] = acc[r] + bb;
}

// ---------------------------------------------------------------------------
// Host launcher
// ---------------------------------------------------------------------------
extern "C" void kernel_launch(void* const* d_in, const int* in_sizes, int n_in,
                              void* d_out, int out_size, void* d_ws, size_t ws_size,
                              hipStream_t stream) {
    const float* x  = (const float*)d_in[0];
    const float* WQ = (const float*)d_in[1];
    const float* WK = (const float*)d_in[2];
    const float* WV = (const float*)d_in[3];
    const float* WO = (const float*)d_in[4];
    const float* bQ = (const float*)d_in[5];
    const float* bK = (const float*)d_in[6];
    const float* bV = (const float*)d_in[7];
    const float* bO = (const float*)d_in[8];
    float* out = (float*)d_out;

    // workspace layout (all f16)
    char* ws = (char*)d_ws;
    const size_t SZ_XH = (size_t)ROWS * DMODEL * 2;          // 16 MB
    const size_t SZ_W  = (size_t)HEADS * DMODEL * DHEAD * 2; //  2 MB each
    const size_t SZ_WO = (size_t)DMODEL * DMODEL * 2;        //  2 MB
    const size_t SZ_QKV= (size_t)BATCH * HEADS * SEQ * DHEAD * 2; // 16 MB each
    _Float16* xh  = (_Float16*)(ws);
    _Float16* wqT = (_Float16*)(ws + SZ_XH);
    _Float16* wkT = (_Float16*)(ws + SZ_XH + SZ_W);
    _Float16* wvT = (_Float16*)(ws + SZ_XH + 2 * SZ_W);
    _Float16* woT = (_Float16*)(ws + SZ_XH + 3 * SZ_W);
    _Float16* q   = (_Float16*)(ws + SZ_XH + 3 * SZ_W + SZ_WO);
    _Float16* k   = (_Float16*)(ws + SZ_XH + 3 * SZ_W + SZ_WO + SZ_QKV);
    _Float16* vT  = (_Float16*)(ws + SZ_XH + 3 * SZ_W + SZ_WO + 2 * SZ_QKV);
    _Float16* z   = (_Float16*)(ws + SZ_XH + 3 * SZ_W + SZ_WO + 3 * SZ_QKV);

    cast_kernel<<<512, 256, 0, stream>>>(x, WQ, WK, WV, WO,
                                         xh, wqT, wkT, wvT, woT);

    // 512 row tiles * 16 heads * 3 matrices = 24576 waves / 4 per block
    qkv_kernel<<<6144, 128, 0, stream>>>(xh, wqT, wkT, wvT, bQ, bK, bV, q, k, vT);

    // 4 * 16 * 128 q-tiles = 8192 waves / 4 per block
    attn_kernel<<<2048, 128, 0, stream>>>(q, k, vT, z);

    // 512 row tiles * 64 col tiles = 32768 waves / 4 per block
    oproj_kernel<<<8192, 128, 0, stream>>>(z, woT, bO, out);
}
